// MolDiscriminator_54211077210639
// MI455X (gfx1250) — compile-verified, matrix operand order checked by reading, not run on hardware
//
#include <hip/hip_runtime.h>

// ---------------------------------------------------------------------------
// MolDiscriminator forward for MI455X (gfx1250, wave32, WMMA f16->f32).
//   B=32, K(bond)=5, N=512, G0=128, G1=64, F1=128, F2=64
// Pipeline:
//   k_prep   : build f16 transposed/padded operands (featTp, Wg0pT, Wg1T)
//   k_layer1 : agg = adj@featPad (WMMA) -> x1 = relu(agg@Wg0p) (WMMA, LDS)
//              -> R[b,k] = x1@Wg1[k] (WMMA) stored transposed (RT)
//   k_layer2 : x2 = relu(sum_k adj@R[b,k]) (WMMA) + per-block column sums
//   k_head   : mean over atoms + 3-layer FC head -> out [B,1] fp32
// Precision: f16 inputs, fp32 WMMA accumulation (adj in [0,1) fits f16).
// ---------------------------------------------------------------------------

typedef __attribute__((ext_vector_type(4)))  float    v4f;
typedef __attribute__((ext_vector_type(8)))  float    v8f;
typedef __attribute__((ext_vector_type(8)))  _Float16 v8h;
typedef __attribute__((ext_vector_type(16))) _Float16 v16h;

#define B_   32
#define K_   5
#define N_   512
#define G0_  128
#define G1_  64

static __device__ __forceinline__ int lane_id() { return (int)(threadIdx.x & 31u); }

static __device__ __forceinline__ v16h cat8(v8h lo, v8h hi) {
  return __builtin_shufflevector(lo, hi, 0,1,2,3,4,5,6,7,8,9,10,11,12,13,14,15);
}

static __device__ __forceinline__ v8f vzero8() {
  v8f z;
#pragma unroll
  for (int i = 0; i < 8; i++) z[i] = 0.f;
  return z;
}

// 16x32 f16 fragment ("row striped across VGPRs within a lane", ISA 7.12.2):
// lane L: row r=L&15, half hh=L>>4; halfs 0..7 = [r][k0+8hh .. +7],
// halfs 8..15 = [r][k0+16+8hh .. +7]. Used for A, and for B via an N-major
// (transposed) copy of the B matrix (symmetric stripe layout).
static __device__ __forceinline__ v16h frag_f16(const _Float16* base, int stride, int k0) {
  int l = lane_id();
  int r = l & 15, hh = l >> 4;
  const _Float16* p = base + (size_t)r * stride + k0 + 8 * hh;
  v8h lo = *(const v8h*)p;
  v8h hi = *(const v8h*)(p + 16);
  return cat8(lo, hi);
}

// Same fragment, sourced from row-major fp32 global memory with on-the-fly cvt.
static __device__ __forceinline__ v16h frag_f32cvt(const float* base, int stride, int k0) {
  int l = lane_id();
  int r = l & 15, hh = l >> 4;
  const float* p = base + (size_t)r * stride + k0 + 8 * hh;
  v4f a0 = *(const v4f*)(p);
  v4f a1 = *(const v4f*)(p + 4);
  v4f b0 = *(const v4f*)(p + 16);
  v4f b1 = *(const v4f*)(p + 20);
  v8h lo, hi;
#pragma unroll
  for (int i = 0; i < 4; i++) {
    lo[i]     = (_Float16)a0[i];
    lo[i + 4] = (_Float16)a1[i];
    hi[i]     = (_Float16)b0[i];
    hi[i + 4] = (_Float16)b1[i];
  }
  return cat8(lo, hi);
}

static __device__ __forceinline__ v8f wmma_f16(v16h a, v16h b, v8f c) {
  // D = A(16x32) * B(32x16) + C, f32 accumulate
  return __builtin_amdgcn_wmma_f32_16x16x32_f16(false, a, false, b, (short)0, c, false, false);
}

// C tile (lane L: col n=L&15, rows M=v+8*(L>>4)) -> f16 row-major [m][n]
static __device__ __forceinline__ void store_tile_f16(_Float16* base, int stride, v8f c, bool relu) {
  int l = lane_id();
  int n = l & 15, hh = l >> 4;
#pragma unroll
  for (int v = 0; v < 8; v++) {
    float x = c[v];
    if (relu) x = fmaxf(x, 0.f);
    base[(size_t)(v + 8 * hh) * stride + n] = (_Float16)x;
  }
}

// C tile -> transposed f16 storage [n][m]; lane's 8 M-values are contiguous
// (single 16-byte store per lane).
static __device__ __forceinline__ void store_tile_tr(_Float16* base, int stride, v8f c) {
  int l = lane_id();
  int n = l & 15, hh = l >> 4;
  v8h o;
#pragma unroll
  for (int v = 0; v < 8; v++) o[v] = (_Float16)c[v];
  *(v8h*)(base + (size_t)n * stride + 8 * hh) = o;
}

// ---------------------------------------------------------------------------
// K0: build f16 operands.
//   featTp [B][16][512]  : feat transposed, bond-feature dim padded 5->16 w/ 0
//   Wg0pT  [128][96]     : Wg0[k][f][o] at col kf=k*16+f (padded 80->96 w/ 0)
//   Wg1T   [5][64][128]  : Wg1[k] transposed (o2-major)
// ---------------------------------------------------------------------------
__global__ __launch_bounds__(256) void k_prep(const float* __restrict__ feat,
                                              const float* __restrict__ Wg0,
                                              const float* __restrict__ Wg1,
                                              _Float16* __restrict__ featTp,
                                              _Float16* __restrict__ Wg0pT,
                                              _Float16* __restrict__ Wg1T) {
  int stride = gridDim.x * blockDim.x;
  int idx = blockIdx.x * blockDim.x + threadIdx.x;
  for (int i = idx; i < B_ * 16 * N_; i += stride) {
    int b = i >> 13, f = (i >> 9) & 15, m = i & 511;
    float v = (f < K_) ? feat[((size_t)b * N_ + m) * K_ + f] : 0.f;
    featTp[i] = (_Float16)v;
  }
  for (int i = idx; i < G0_ * 96; i += stride) {
    int o = i / 96, kf = i % 96, k = kf >> 4, f = kf & 15;
    float v = (k < K_ && f < K_) ? Wg0[((size_t)k * K_ + f) * G0_ + o] : 0.f;
    Wg0pT[i] = (_Float16)v;
  }
  for (int i = idx; i < K_ * G1_ * G0_; i += stride) {
    int k = i >> 13, r = i & 8191, o2 = r >> 7, o1 = r & 127;
    Wg1T[i] = (_Float16)Wg1[((size_t)k * G0_ + o1) * G1_ + o2];
  }
}

// ---------------------------------------------------------------------------
// K1: fused layer-1 rgconv + layer-2 pre-projection.
// Block = 4 waves, each wave owns 16 atom-rows; block tile = 64 rows.
// Grid = B * (N/64) = 256 blocks.
// ---------------------------------------------------------------------------
__global__ __launch_bounds__(128) void k_layer1(const float* __restrict__ adj,
                                                const _Float16* __restrict__ featTp,
                                                const _Float16* __restrict__ Wg0pT,
                                                const _Float16* __restrict__ Wg1T,
                                                _Float16* __restrict__ RT) {
  __shared__ __align__(16) _Float16 sAgg[64 * 104]; // [n][kf] stride 104 (pad)
  __shared__ __align__(16) _Float16 sX1[64 * 136];  // [n][o]  stride 136 (pad)
  int b = blockIdx.x >> 3, mt = blockIdx.x & 7;
  int wave = (int)(threadIdx.x >> 5);
  int n0 = mt * 64 + wave * 16;
  _Float16* aggW = sAgg + wave * 16 * 104;
  _Float16* x1W  = sX1 + wave * 16 * 136;

  // zero this wave's agg rows (covers the kf=80..95 zero pad)
  for (int i = lane_id(); i < 16 * 104; i += 32) aggW[i] = (_Float16)0.f;

  const _Float16* featB = featTp + (size_t)b * 16 * N_;

  // Stage 1: agg[n, k*16+f] = sum_m adj[b,k,n,m] * featPad[b,m,f]
  for (int k = 0; k < K_; k++) {
    const float* A = adj + (((size_t)(b * K_ + k)) * N_ + n0) * N_;
    v8f acc = vzero8();
    for (int k0 = 0; k0 < N_; k0 += 32) {
      v16h af = frag_f32cvt(A, N_, k0);
      v16h bf = frag_f16(featB, N_, k0);
      acc = wmma_f16(af, bf, acc);
    }
    store_tile_f16(aggW + k * 16, 104, acc, false);
  }

  // Stage 2: x1[n,o] = relu(sum_kf agg[n,kf] * Wg0p[kf,o])  (K-dim 96)
  for (int ot = 0; ot < 8; ot++) {
    v8f acc = vzero8();
#pragma unroll
    for (int kf0 = 0; kf0 < 96; kf0 += 32) {
      v16h af = frag_f16(aggW, 104, kf0);
      v16h bf = frag_f16(Wg0pT + ot * 16 * 96, 96, kf0);
      acc = wmma_f16(af, bf, acc);
    }
    store_tile_f16(x1W + ot * 16, 136, acc, true);
  }

  // Stage 3: R_k = x1 @ Wg1[k] -> RT[b][k][o2][n] (n-contiguous)
  for (int k = 0; k < K_; k++) {
    const _Float16* Bk = Wg1T + (size_t)k * G1_ * G0_;
    for (int ot = 0; ot < 4; ot++) {
      v8f acc = vzero8();
#pragma unroll
      for (int k0 = 0; k0 < G0_; k0 += 32) {
        v16h af = frag_f16(x1W, 136, k0);
        v16h bf = frag_f16(Bk + ot * 16 * G0_, G0_, k0);
        acc = wmma_f16(af, bf, acc);
      }
      _Float16* out = RT + (((size_t)(b * K_ + k)) * G1_ + ot * 16) * N_ + n0;
      store_tile_tr(out, N_, acc);
    }
  }
}

// ---------------------------------------------------------------------------
// K2: x2[n,o2] = relu(sum_k adj[b,k] @ R[b,k]); emit per-block column sums
// (deterministic readout; no float atomics). Grid = 256 blocks of 4 waves.
// ---------------------------------------------------------------------------
__global__ __launch_bounds__(128) void k_layer2(const float* __restrict__ adj,
                                                const _Float16* __restrict__ RT,
                                                float* __restrict__ psum) {
  __shared__ float sRed[4][64];
  int b = blockIdx.x >> 3, mt = blockIdx.x & 7;
  int wave = (int)(threadIdx.x >> 5);
  int n0 = mt * 64 + wave * 16;

  v8f acc[4];
#pragma unroll
  for (int ot = 0; ot < 4; ot++) acc[ot] = vzero8();

  for (int k = 0; k < K_; k++) {
    const float* A = adj + (((size_t)(b * K_ + k)) * N_ + n0) * N_;
    const _Float16* Bk = RT + ((size_t)(b * K_ + k)) * G1_ * N_;
    for (int k0 = 0; k0 < N_; k0 += 32) {
      if (k0 + 160 <= N_) { // stream-ahead prefetch of this wave's adj rows
        const float* pp = A + (size_t)(lane_id() & 15) * N_ + k0 + 128;
        __builtin_prefetch(pp, 0, 1);
      }
      v16h af = frag_f32cvt(A, N_, k0);
#pragma unroll
      for (int ot = 0; ot < 4; ot++) {
        v16h bf = frag_f16(Bk + ot * 16 * N_, N_, k0);
        acc[ot] = wmma_f16(af, bf, acc[ot]);
      }
    }
  }

  // ReLU + column partial sums over this wave's 16 rows (wave32 shuffle)
  int l = lane_id();
  int n = l & 15, hh = l >> 4;
#pragma unroll
  for (int ot = 0; ot < 4; ot++) {
    float s = 0.f;
#pragma unroll
    for (int v = 0; v < 8; v++) s += fmaxf(acc[ot][v], 0.f);
    s += __shfl_xor(s, 16); // combine M0..7 with M8..15 halves of the column
    if (hh == 0) sRed[wave][ot * 16 + n] = s;
  }
  __syncthreads();
  if (threadIdx.x < 64) {
    float t = sRed[0][threadIdx.x] + sRed[1][threadIdx.x] +
              sRed[2][threadIdx.x] + sRed[3][threadIdx.x];
    psum[((size_t)b * 8 + mt) * 64 + threadIdx.x] = t;
  }
}

// ---------------------------------------------------------------------------
// K3: graph readout (mean over atoms) + FC head. Grid = B blocks.
// ---------------------------------------------------------------------------
__global__ __launch_bounds__(128) void k_head(const float* __restrict__ psum,
                                              const float* __restrict__ W1, const float* __restrict__ b1,
                                              const float* __restrict__ W2, const float* __restrict__ b2,
                                              const float* __restrict__ W3, const float* __restrict__ b3,
                                              float* __restrict__ out) {
  __shared__ float x[64], h1[128], h2[64];
  int b = blockIdx.x, t = (int)threadIdx.x;
  if (t < 64) {
    float s = 0.f;
#pragma unroll
    for (int mt = 0; mt < 8; mt++) s += psum[((size_t)b * 8 + mt) * 64 + t];
    x[t] = s * (1.f / 512.f);
  }
  __syncthreads();
  {
    float s = b1[t];
    for (int i = 0; i < 64; i++) s += x[i] * W1[i * 128 + t];
    h1[t] = fmaxf(s, 0.f);
  }
  __syncthreads();
  if (t < 64) {
    float s = b2[t];
    for (int i = 0; i < 128; i++) s += h1[i] * W2[i * 64 + t];
    h2[t] = fmaxf(s, 0.f);
  }
  __syncthreads();
  if (t == 0) {
    float s = b3[0];
    for (int i = 0; i < 64; i++) s += h2[i] * W3[i];
    out[b] = s;
  }
}

// ---------------------------------------------------------------------------
// Host launcher. Workspace layout (bytes, 256-aligned):
//   0        featTp  (512 KB)
//   524288   Wg0pT   (24 KB)
//   548864   Wg1T    (80 KB)
//   630784   RT      (10.0 MB)  [B][K][64][512] f16
//   11116544 psum    (64 KB)    [B][8][64] f32
// ---------------------------------------------------------------------------
extern "C" void kernel_launch(void* const* d_in, const int* in_sizes, int n_in,
                              void* d_out, int out_size, void* d_ws, size_t ws_size,
                              hipStream_t stream) {
  const float* adj  = (const float*)d_in[0];
  const float* feat = (const float*)d_in[1];
  const float* Wg0  = (const float*)d_in[2];
  const float* Wg1  = (const float*)d_in[3];
  const float* W1   = (const float*)d_in[4];
  const float* b1   = (const float*)d_in[5];
  const float* W2   = (const float*)d_in[6];
  const float* b2   = (const float*)d_in[7];
  const float* W3   = (const float*)d_in[8];
  const float* b3   = (const float*)d_in[9];
  float* out = (float*)d_out;

  char* ws = (char*)d_ws;
  _Float16* featTp = (_Float16*)(ws + 0);
  _Float16* Wg0pT  = (_Float16*)(ws + 524288);
  _Float16* Wg1T   = (_Float16*)(ws + 548864);
  _Float16* RT     = (_Float16*)(ws + 630784);
  float*    psum   = (float*)(ws + 11116544);

  k_prep<<<256, 256, 0, stream>>>(feat, Wg0, Wg1, featTp, Wg0pT, Wg1T);
  k_layer1<<<256, 128, 0, stream>>>(adj, featTp, Wg0pT, Wg1T, RT);
  k_layer2<<<256, 128, 0, stream>>>(adj, RT, psum);
  k_head<<<32, 128, 0, stream>>>(psum, W1, b1, W2, b2, W3, b3, out);

  (void)in_sizes; (void)n_in; (void)out_size; (void)ws_size;
}
